// WeightedSum_33036888441237
// MI455X (gfx1250) — matrix operand order repaired and checked by reading, not compiled, not run on hardware
//
#include <hip/hip_runtime.h>
#include <hip/hip_bf16.h>

// ---------------------------------------------------------------------------
// CDNA5 fused kernel for the Cartesian-tensor weighted-sum reference.
//   Phase A: channel-mixing GEMMs via v_wmma_f32_16x16x32_bf16 with a
//            bf16 hi/lo split (3 WMMAs per K-step) for ~fp32 accuracy.
//            Job schedule is fully unrolled from constexpr tables so the
//            hot loop has no table loads and no branches.
//   Phase B: per-(edge,channel) D=3 contraction paths + pw-weighted sums,
//            done on f32 values staged in LDS, outputs held in registers.
// ---------------------------------------------------------------------------

typedef __attribute__((ext_vector_type(8)))  __bf16 v8bf;
typedef __attribute__((ext_vector_type(16))) __bf16 v16bf;
typedef __attribute__((ext_vector_type(8)))  float  v8f;

#define NC        128                    // channels
#define EBLK      16                     // edges per workgroup (WMMA M)
#define NSLOT     19                     // max mixed slots staged per group
#define MIX_STRIDE (EBLK*NC)             // 2048 floats per slot
#define LDS_MIX_BYTES (NSLOT*MIX_STRIDE*4)       // 155648
#define LDS_A_ELEMS   (13*EBLK*NC)               // 26624 bf16 per (hi|lo)
#define LDS_A_BYTES   (LDS_A_ELEMS*2)            // 53248
#define LDS_TOTAL     (LDS_MIX_BYTES + 2*LDS_A_BYTES)  // 262144 (<320KB/WGP)

// 15 mixed tensors, global mix index mm:
//  0 s0   = cm_0[0]  a0      5 v1   = cm_1[0]  a1     10 M2   = cm_2[0]  a2
//  1 s00a = cm_00[0] a0      6 v01  = cm_01[1] a1     11 M02  = cm_02[1] a2
//  2 s00b = cm_00[1] a0      7 v11a = cm_11[0] a1     12 M12  = cm_12[1] a2
//  3 s01  = cm_01[0] a0      8 v11b = cm_11[1] a1     13 M22a = cm_22[0] a2
//  4 s02  = cm_02[0] a0      9 v12  = cm_12[0] a1     14 M22b = cm_22[1] a2

// groups of co-used mixed tensors (bounded LDS staging):
//  G1: s0,s00a,s00b,s01,v1,v01,M2       (19 slots)
//  G2: s02,M02,v11a,v11b                (16 slots)
//  G3: v12,M12                          (12 slots)
//  G4: M22a,M22b                        (18 slots)
constexpr int k_gstart[5] = {0, 19, 35, 47, 65};

constexpr int k_mm[65] = {
  0,1,2,3, 5,5,5, 6,6,6, 10,10,10,10,10,10,10,10,10,
  4, 11,11,11,11,11,11,11,11,11, 7,7,7, 8,8,8,
  9,9,9, 12,12,12,12,12,12,12,12,12,
  13,13,13,13,13,13,13,13,13, 14,14,14,14,14,14,14,14,14 };

// plane: 0 = a0; 1..3 = a1 comp d; 4..12 = a2 comp (flattened 3x3)
constexpr int k_pl[65] = {
  0,0,0,0, 1,2,3, 1,2,3, 4,5,6,7,8,9,10,11,12,
  0, 4,5,6,7,8,9,10,11,12, 1,2,3, 1,2,3,
  1,2,3, 4,5,6,7,8,9,10,11,12,
  4,5,6,7,8,9,10,11,12, 4,5,6,7,8,9,10,11,12 };

constexpr int k_slot[65] = {
  0,1,2,3, 4,5,6, 7,8,9, 10,11,12,13,14,15,16,17,18,
  0, 1,2,3,4,5,6,7,8,9, 10,11,12, 13,14,15,
  0,1,2, 3,4,5,6,7,8,9,10,11,
  0,1,2,3,4,5,6,7,8, 9,10,11,12,13,14,15,16,17 };

constexpr int k_newB[65] = {
  1,1,1,1, 1,0,0, 1,0,0, 1,0,0,0,0,0,0,0,0,
  1, 1,0,0,0,0,0,0,0,0, 1,0,0, 1,0,0,
  1,0,0, 1,0,0,0,0,0,0,0,0,
  1,0,0,0,0,0,0,0,0, 1,0,0,0,0,0,0,0,0 };

// source (cm buffer index within d_in[3..11], row) for each mm
__device__ __constant__ unsigned char c_srcIdx[15] = {0,3,3,4,5,1,4,6,6,7,2,5,7,8,8};
__device__ __constant__ unsigned char c_srcRow[15] = {0,0,1,0,0,0,1,0,1,0,0,1,1,0,1};

__device__ __forceinline__ v16bf cat16(v8bf a, v8bf b) {
  return __builtin_shufflevector(a, b, 0,1,2,3,4,5,6,7,8,9,10,11,12,13,14,15);
}

__device__ __forceinline__ void cvt_store(__bf16* hi, __bf16* lo, int idx, float x) {
  __bf16 h = (__bf16)x;
  hi[idx] = h;
  lo[idx] = (__bf16)(x - (float)h);
}

// ---------------------------------------------------------------------------
// pre-kernel: pack the 15 cm matrices into bf16 hi/lo planes in d_ws
// layout: B[mm][i][j], row-major, j contiguous (WMMA B fragments read 16
// contiguous j per lane: lane n fixes out-channel i, covers K=j run).
// ---------------------------------------------------------------------------
struct CmPtrs { const float* p[9]; };

__global__ __launch_bounds__(256)
void pack_cm_kernel(CmPtrs cms, __bf16* __restrict__ bhi, __bf16* __restrict__ blo)
{
  const int f = blockIdx.x * 256 + threadIdx.x;
  if (f >= 15 * NC * NC) return;
  const int mm = f / (NC * NC);
  const int r  = f - mm * (NC * NC);
  const float x = cms.p[c_srcIdx[mm]][(size_t)c_srcRow[mm] * NC * NC + r];
  const __bf16 h = (__bf16)x;
  bhi[f] = h;
  blo[f] = (__bf16)(x - (float)h);
}

// ---------------------------------------------------------------------------
// Phase A: one mix group. Each wave owns out-channel tile [wave*16, wave*16+16).
// Per tile job: 4 K-steps x 3 hi/lo-term WMMAs, acc stored as f32 to ldsMix.
// Fully unrolled: all job indices fold to immediates.
// ---------------------------------------------------------------------------
template <int G>
__device__ __forceinline__ void phaseA(
    const __bf16* __restrict__ Bhi, const __bf16* __restrict__ Blo,
    const __bf16* ldsAhi, const __bf16* ldsAlo, float* ldsMix)
{
  const int lane = threadIdx.x & 31;
  const int wave = threadIdx.x >> 5;
  const int h    = lane >> 4;      // half-wave (K-group select)
  const int mn   = lane & 15;      // M for A-frag, N for B-frag
  const int icol = wave * 16 + mn; // global output channel

  v16bf bh[4] = {}, bl[4] = {};
#pragma unroll
  for (int idx = k_gstart[G]; idx < k_gstart[G + 1]; ++idx) {
    constexpr_fold:;
    const int mm = k_mm[idx], pl = k_pl[idx], slot = k_slot[idx];
    if (k_newB[idx]) {  // compile-time constant after unroll
      const __bf16* bp = Bhi + ((size_t)(mm * NC + icol)) * NC + h * 16;
      const __bf16* lp = Blo + ((size_t)(mm * NC + icol)) * NC + h * 16;
#pragma unroll
      for (int kk = 0; kk < 4; ++kk) {
        const v8bf* b8 = (const v8bf*)(bp + kk * 32);
        bh[kk] = cat16(b8[0], b8[1]);
        const v8bf* l8 = (const v8bf*)(lp + kk * 32);
        bl[kk] = cat16(l8[0], l8[1]);
      }
    }
    v8f acc = {0.f,0.f,0.f,0.f,0.f,0.f,0.f,0.f};
    const __bf16* ah = ldsAhi + (pl * EBLK + mn) * NC + h * 16;
    const __bf16* al = ldsAlo + (pl * EBLK + mn) * NC + h * 16;
#pragma unroll
    for (int kk = 0; kk < 4; ++kk) {
      // A frag, contiguous K order chosen here: lane covers K run [h*16,h*16+16) (matches B)
      const v8bf ah0 = *(const v8bf*)(ah + kk * 32);
      const v8bf ah1 = *(const v8bf*)(ah + kk * 32 + 8);
      const v16bf Ahi = cat16(ah0, ah1);
      const v8bf al0 = *(const v8bf*)(al + kk * 32);
      const v8bf al1 = *(const v8bf*)(al + kk * 32 + 8);
      const v16bf Alo = cat16(al0, al1);
      acc = __builtin_amdgcn_wmma_f32_16x16x32_bf16(false, Ahi, false, bh[kk], (short)0, acc, false, false);
      acc = __builtin_amdgcn_wmma_f32_16x16x32_bf16(false, Ahi, false, bl[kk], (short)0, acc, false, false);
      acc = __builtin_amdgcn_wmma_f32_16x16x32_bf16(false, Alo, false, bh[kk], (short)0, acc, false, false);
    }
    // C/D layout: lane(l), vgpr v -> (M = v + 8*(l>>4), N = l&15)
    float* mp = ldsMix + slot * MIX_STRIDE + h * 8 * NC + icol;
#pragma unroll
    for (int v = 0; v < 8; ++v) mp[v * NC] = acc[v];
  }
}

// ---------------------------------------------------------------------------
// fused main kernel: 256 threads (8 wave32), one 16-edge block per WG
// ---------------------------------------------------------------------------
__global__ __launch_bounds__(256)
void fused_kernel(const float* __restrict__ a0, const float* __restrict__ a1,
                  const float* __restrict__ a2,
                  const __bf16* __restrict__ Bhi, const __bf16* __restrict__ Blo,
                  const float* __restrict__ pw0, const float* __restrict__ pw1,
                  const float* __restrict__ pw2,
                  float* __restrict__ out, int E)
{
  extern __shared__ char smem[];
  float*  ldsMix = (float*)smem;
  __bf16* ldsAhi = (__bf16*)(smem + LDS_MIX_BYTES);
  __bf16* ldsAlo = (__bf16*)(smem + LDS_MIX_BYTES + LDS_A_BYTES);

  const int  tid   = threadIdx.x;
  const long ebase = (long)blockIdx.x * EBLK;

  // ---- Phase 0: stage edge-block inputs as bf16 hi/lo planes in LDS ----
  for (int pair = tid; pair < EBLK * NC; pair += 256) {
    const int  e = pair >> 7, j = pair & 127;
    const long g = (ebase + e) * NC + j;
    cvt_store(ldsAhi, ldsAlo, (0 * EBLK + e) * NC + j, a0[g]);
    const float* p1 = a1 + g * 3;
#pragma unroll
    for (int d = 0; d < 3; ++d)
      cvt_store(ldsAhi, ldsAlo, ((1 + d) * EBLK + e) * NC + j, p1[d]);
    const float* p2 = a2 + g * 9;
#pragma unroll
    for (int q = 0; q < 9; ++q)
      cvt_store(ldsAhi, ldsAlo, ((4 + q) * EBLK + e) * NC + j, p2[q]);
  }

  // Phase-B identity: thread -> channel c, edge half eh (8 edges each)
  const int c  = tid & 127;
  const int eh = tid >> 7;

  float acc0[8];
  float acc1[8][3];
  float acc2[8][9];

  __syncthreads();

  // ================= Group 1: s0,s00a,s00b,s01,v1,v01,M2 =================
  phaseA<0>(Bhi, Blo, ldsAhi, ldsAlo, ldsMix);
  __syncthreads();
  {
    const float w00 = pw0[0*NC+c], w01 = pw0[1*NC+c], w02 = pw0[2*NC+c];
    const float w10 = pw1[0*NC+c], w11 = pw1[1*NC+c];
    const float w20 = pw2[0*NC+c];
#pragma unroll
    for (int k = 0; k < 8; ++k) {
      const int e = eh * 8 + k;
      const float* mx = ldsMix + e * NC + c;
      const float s0  = mx[0*MIX_STRIDE], sA = mx[1*MIX_STRIDE];
      const float sB  = mx[2*MIX_STRIDE], s01 = mx[3*MIX_STRIDE];
      float v1d[3], v01d[3], M2[9];
#pragma unroll
      for (int d = 0; d < 3; ++d) { v1d[d] = mx[(4+d)*MIX_STRIDE]; v01d[d] = mx[(7+d)*MIX_STRIDE]; }
#pragma unroll
      for (int q = 0; q < 9; ++q) M2[q] = mx[(10+q)*MIX_STRIDE];
      const float trM2 = M2[0] + M2[4] + M2[8];
      acc0[k] = w00 * s0 + w01 * trM2 + w02 * sA * sB;
#pragma unroll
      for (int d = 0; d < 3; ++d) acc1[k][d] = w10 * v1d[d] + w11 * s01 * v01d[d];
#pragma unroll
      for (int q = 0; q < 9; ++q) acc2[k][q] = w20 * M2[q];
    }
  }
  __syncthreads();

  // ================= Group 2: s02,M02,v11a,v11b =================
  phaseA<1>(Bhi, Blo, ldsAhi, ldsAlo, ldsMix);
  __syncthreads();
  {
    const float w03 = pw0[3*NC+c], w04 = pw0[4*NC+c];
    const float w21 = pw2[1*NC+c], w22 = pw2[2*NC+c];
#pragma unroll
    for (int k = 0; k < 8; ++k) {
      const int e = eh * 8 + k;
      const float* mx = ldsMix + e * NC + c;
      const float s02 = mx[0*MIX_STRIDE];
      float M02[9], va[3], vb[3];
#pragma unroll
      for (int q = 0; q < 9; ++q) M02[q] = mx[(1+q)*MIX_STRIDE];
#pragma unroll
      for (int d = 0; d < 3; ++d) { va[d] = mx[(10+d)*MIX_STRIDE]; vb[d] = mx[(13+d)*MIX_STRIDE]; }
      const float trM02 = M02[0] + M02[4] + M02[8];
      const float dotv  = va[0]*vb[0] + va[1]*vb[1] + va[2]*vb[2];
      acc0[k] += w03 * s02 * trM02 + w04 * dotv;
#pragma unroll
      for (int d = 0; d < 3; ++d)
#pragma unroll
        for (int e2 = 0; e2 < 3; ++e2)
          acc2[k][d*3+e2] += w21 * s02 * M02[d*3+e2] + w22 * va[d] * vb[e2];
    }
  }
  __syncthreads();

  // ================= Group 3: v12,M12 =================
  phaseA<2>(Bhi, Blo, ldsAhi, ldsAlo, ldsMix);
  __syncthreads();
  {
    const float w12 = pw1[2*NC+c], w13 = pw1[3*NC+c], w14 = pw1[4*NC+c];
#pragma unroll
    for (int k = 0; k < 8; ++k) {
      const int e = eh * 8 + k;
      const float* mx = ldsMix + e * NC + c;
      float v[3], M[9];
#pragma unroll
      for (int d = 0; d < 3; ++d) v[d] = mx[d*MIX_STRIDE];
#pragma unroll
      for (int q = 0; q < 9; ++q) M[q] = mx[(3+q)*MIX_STRIDE];
      const float trM = M[0] + M[4] + M[8];
#pragma unroll
      for (int d = 0; d < 3; ++d) {
        const float q2 = v[0]*M[0*3+d] + v[1]*M[1*3+d] + v[2]*M[2*3+d]; // v^T M
        const float q3 = M[d*3+0]*v[0] + M[d*3+1]*v[1] + M[d*3+2]*v[2]; // M v
        acc1[k][d] += w12 * q2 + w13 * q3 + w14 * v[d] * trM;
      }
    }
  }
  __syncthreads();

  // ================= Group 4: M22a,M22b =================
  phaseA<3>(Bhi, Blo, ldsAhi, ldsAlo, ldsMix);
  __syncthreads();
  {
    const float w05 = pw0[5*NC+c], w06 = pw0[6*NC+c], w07 = pw0[7*NC+c];
    const float w23 = pw2[3*NC+c], w24 = pw2[4*NC+c], w25 = pw2[5*NC+c];
    const float w26 = pw2[6*NC+c], w27 = pw2[7*NC+c], w28 = pw2[8*NC+c];
#pragma unroll
    for (int k = 0; k < 8; ++k) {
      const int e = eh * 8 + k;
      const float* mx = ldsMix + e * NC + c;
      float A[9], B[9];
#pragma unroll
      for (int q = 0; q < 9; ++q) { A[q] = mx[q*MIX_STRIDE]; B[q] = mx[(9+q)*MIX_STRIDE]; }
      const float tA = A[0] + A[4] + A[8];
      const float tB = B[0] + B[4] + B[8];
      float s6 = 0.f, s7 = 0.f;
#pragma unroll
      for (int d = 0; d < 3; ++d)
#pragma unroll
        for (int e2 = 0; e2 < 3; ++e2) {
          s6 += A[d*3+e2] * B[d*3+e2];
          s7 += A[d*3+e2] * B[e2*3+d];
        }
      acc0[k] += w05 * tA * tB + w06 * s6 + w07 * s7;
#pragma unroll
      for (int d = 0; d < 3; ++d)
#pragma unroll
        for (int e2 = 0; e2 < 3; ++e2) {
          float r4 = 0.f, r5 = 0.f, r6 = 0.f, r7 = 0.f;
#pragma unroll
          for (int cc = 0; cc < 3; ++cc) {
            r4 += A[cc*3+d]  * B[cc*3+e2];  // A^T B
            r5 += A[cc*3+d]  * B[e2*3+cc];  // A^T B^T
            r6 += A[d*3+cc]  * B[cc*3+e2];  // A B
            r7 += A[d*3+cc]  * B[e2*3+cc];  // A B^T
          }
          acc2[k][d*3+e2] += w23 * tA * B[d*3+e2] + w24 * r4 + w25 * r5
                           + w26 * r6 + w27 * r7 + w28 * A[d*3+e2] * tB;
        }
    }
  }

  // ---- write outputs (each (e,c) owned by exactly one thread) ----
  {
    const long EC = (long)E * NC;
    float* out0 = out;
    float* out1 = out + EC;
    float* out2 = out + EC * 4;  // EC + 3*EC
#pragma unroll
    for (int k = 0; k < 8; ++k) {
      const int  e = eh * 8 + k;
      const long g = (ebase + e) * NC + c;
      out0[g] = acc0[k];
#pragma unroll
      for (int d = 0; d < 3; ++d) out1[g*3 + d] = acc1[k][d];
#pragma unroll
      for (int q = 0; q < 9; ++q) out2[g*9 + q] = acc2[k][q];
    }
  }
}

// ---------------------------------------------------------------------------
extern "C" void kernel_launch(void* const* d_in, const int* in_sizes, int n_in,
                              void* d_out, int out_size, void* d_ws, size_t ws_size,
                              hipStream_t stream)
{
  (void)n_in; (void)out_size; (void)ws_size;
  const float* a0 = (const float*)d_in[0];
  const float* a1 = (const float*)d_in[1];
  const float* a2 = (const float*)d_in[2];
  CmPtrs cms;
  for (int i = 0; i < 9; ++i) cms.p[i] = (const float*)d_in[3 + i];
  const float* pw0 = (const float*)d_in[12];
  const float* pw1 = (const float*)d_in[13];
  const float* pw2 = (const float*)d_in[14];

  __bf16* bhi = (__bf16*)d_ws;
  __bf16* blo = bhi + 15 * NC * NC;

  const int E = in_sizes[0] / NC;

  pack_cm_kernel<<<(15 * NC * NC + 255) / 256, 256, 0, stream>>>(cms, bhi, blo);

  hipFuncSetAttribute((const void*)fused_kernel,
                      hipFuncAttributeMaxDynamicSharedMemorySize, LDS_TOTAL);
  fused_kernel<<<E / EBLK, 256, LDS_TOTAL, stream>>>(
      a0, a1, a2, bhi, blo, pw0, pw1, pw2, (float*)d_out, E);
}
